// Encoder_19146964205981
// MI455X (gfx1250) — matrix operand order repaired and checked by reading, not compiled, not run on hardware
//
#include <hip/hip_runtime.h>
#include <hip/hip_bf16.h>
#include <stdint.h>

typedef __attribute__((ext_vector_type(16))) __bf16 v16bf;
typedef __attribute__((ext_vector_type(8)))  __bf16 v8bf;
typedef __attribute__((ext_vector_type(4)))  __bf16 v4bf;
typedef __attribute__((ext_vector_type(8)))  float  v8f;

#define DEVINL static __device__ __forceinline__

DEVINL __bf16 f2b(float f) { return (__bf16)f; }

// ------------------------------------------------------------------ zero (float4)
__global__ void k_zero4(float4* p, size_t n4) {
  size_t i = (size_t)blockIdx.x * blockDim.x + threadIdx.x;
  if (i < n4) p[i] = make_float4(0.f, 0.f, 0.f, 0.f);
}

// ------------------------------------------------------------------ f32 -> bf16 (4-wide)
__global__ void k_cvt_bf16_4(const float4* __restrict__ in, v4bf* __restrict__ out, size_t n4) {
  size_t i = (size_t)blockIdx.x * blockDim.x + threadIdx.x;
  if (i >= n4) return;
  float4 f = in[i];
  v4bf b;
  b[0] = f2b(f.x); b[1] = f2b(f.y); b[2] = f2b(f.z); b[3] = f2b(f.w);
  out[i] = b;
}

// ------------------------------------------------------------------ degree
__global__ void k_degree(const int* __restrict__ dst, float* __restrict__ deg, int E) {
  int e = blockIdx.x * blockDim.x + threadIdx.x;
  if (e < E) atomicAdd(&deg[dst[e]], 1.0f);
}

// ------------------------------------------------------------------ edge gather + scatter-add (float4 per thread)
__global__ void k_scatter4(const int* __restrict__ src, const int* __restrict__ dst,
                           const float* __restrict__ X, float* __restrict__ AGG,
                           int per4, int F, long long total) {
  long long gid = (long long)blockIdx.x * blockDim.x + threadIdx.x;
  if (gid >= total) return;
  int e  = (int)(gid / per4);
  int f4 = (int)(gid - (long long)e * per4);
  int s = src[e], d = dst[e];
  const float4 v = *(const float4*)(X + (size_t)s * F + (size_t)f4 * 4);
  float* a = AGG + (size_t)d * F + (size_t)f4 * 4;
  atomicAdd(a + 0, v.x); atomicAdd(a + 1, v.y);
  atomicAdd(a + 2, v.z); atomicAdd(a + 3, v.w);
}

// ------------------------------------------------------------------ mean divide + bf16 convert (4-wide)
__global__ void k_mean_bf16(const float4* __restrict__ AGG, const float* __restrict__ deg,
                            v4bf* __restrict__ out, int F4, size_t n4) {
  size_t i = (size_t)blockIdx.x * blockDim.x + threadIdx.x;
  if (i >= n4) return;
  int node = (int)(i / F4);
  float inv = 1.0f / fmaxf(deg[node], 1.0f);
  float4 a = AGG[i];
  v4bf b;
  b[0] = f2b(a.x * inv); b[1] = f2b(a.y * inv);
  b[2] = f2b(a.z * inv); b[3] = f2b(a.w * inv);
  out[i] = b;
}

// ------------------------------------------------------------------ WMMA fragment loaders (ISA 7.12.2 layouts)
// A 16x32 bf16: lane L -> row = L%16; lanes 0-15 hold K {0..7,16..23}, lanes 16-31 K {8..15,24..31}
DEVINL v16bf load_afrag(const __bf16* __restrict__ row, int k0, int half) {
  union { v16bf v; v8bf h[2]; } u;
  u.h[0] = *(const v8bf*)(row + k0 + half * 8);        // K = k0 + half*8 .. +8
  u.h[1] = *(const v8bf*)(row + k0 + 16 + half * 8);   // K = k0+16+half*8 .. +8
  return u.v;
}
// B 32x16 bf16 from W[j,k] row-major: lane L -> col j = L%16; lanes 0-15 hold K {0..7,16..23}, lanes 16-31 K {8..15,24..31}
DEVINL v16bf load_bfrag(const __bf16* __restrict__ wrow, int k0, int half) {
  union { v16bf v; v8bf h[2]; } u;
  u.h[0] = *(const v8bf*)(wrow + k0 + half * 8);
  u.h[1] = *(const v8bf*)(wrow + k0 + 16 + half * 8);
  return u.v;
}

// ------------------------------------------------------------------ fused SAGE GEMM: H = Aagg@Wl^T + Aroot@Wr^T + bias
// one wave per 16x16 output tile; K-loop in steps of 32 (bf16), f32 accumulation
__global__ void __launch_bounds__(256) k_sage_gemm(
    const __bf16* __restrict__ Aagg, const __bf16* __restrict__ Aroot,
    const __bf16* __restrict__ Wl,   const __bf16* __restrict__ Wr,
    const float*  __restrict__ bias, float* __restrict__ H,
    int K, int M, int tilesPerRow, int totalTiles)
{
  int wave = (int)(((size_t)blockIdx.x * blockDim.x + threadIdx.x) >> 5);
  if (wave >= totalTiles) return;          // wave-granular: EXEC stays all-ones for WMMA
  int lane = threadIdx.x & 31;
  int trow = wave / tilesPerRow;
  int tcol = wave - trow * tilesPerRow;
  int row0 = trow << 4, col0 = tcol << 4;
  int half = lane >> 4, r = lane & 15;

  const __bf16* aagg  = Aagg  + (size_t)(row0 + r) * K;
  const __bf16* aroot = Aroot + (size_t)(row0 + r) * K;
  const __bf16* bl    = Wl    + (size_t)(col0 + r) * K;
  const __bf16* br    = Wr    + (size_t)(col0 + r) * K;

  v8f acc = {0.f, 0.f, 0.f, 0.f, 0.f, 0.f, 0.f, 0.f};
  for (int k0 = 0; k0 < K; k0 += 32) {
    v16bf a = load_afrag(aagg, k0, half);
    v16bf b = load_bfrag(bl, k0, half);
    acc = __builtin_amdgcn_wmma_f32_16x16x32_bf16(false, a, false, b, (short)0, acc, false, false);
  }
  for (int k0 = 0; k0 < K; k0 += 32) {
    v16bf a = load_afrag(aroot, k0, half);
    v16bf b = load_bfrag(br, k0, half);
    acc = __builtin_amdgcn_wmma_f32_16x16x32_bf16(false, a, false, b, (short)0, acc, false, false);
  }
  // C/D layout: VGPR v -> row = row0 + v + half*8, col = col0 + (lane%16)
  float bv = bias[col0 + r];
  float* outp = H + (size_t)(row0 + half * 8) * M + col0 + r;
#pragma unroll
  for (int v = 0; v < 8; ++v) outp[(size_t)v * M] = acc[v] + bv;
}

// ------------------------------------------------------------------ BN stats: one block per column, LDS tree reduce
__global__ void __launch_bounds__(256) k_bn_stats(const float* __restrict__ H, int Nrows, int M,
                                                  float* __restrict__ mean, float* __restrict__ rsig) {
  __shared__ float ss[256];
  __shared__ float qq[256];
  int col = blockIdx.x;
  float s = 0.f, q = 0.f;
  for (int rw = threadIdx.x; rw < Nrows; rw += blockDim.x) {
    float v = H[(size_t)rw * M + col];
    s += v; q += v * v;
  }
  ss[threadIdx.x] = s; qq[threadIdx.x] = q;
  __syncthreads();
  for (int o = 128; o > 0; o >>= 1) {
    if ((int)threadIdx.x < o) {
      ss[threadIdx.x] += ss[threadIdx.x + o];
      qq[threadIdx.x] += qq[threadIdx.x + o];
    }
    __syncthreads();
  }
  if (threadIdx.x == 0) {
    float mu  = ss[0] / (float)Nrows;
    float var = qq[0] / (float)Nrows - mu * mu;
    mean[col] = mu;
    rsig[col] = rsqrtf(fmaxf(var, 0.f) + 1e-5f);
  }
}

// ------------------------------------------------------------------ BN apply (+optional ReLU), fp32 out + optional bf16 out
__global__ void k_bn_apply4(const float4* H, const float* __restrict__ mean,
                            const float* __restrict__ rsig, const float* __restrict__ g,
                            const float* __restrict__ bt, int M4, size_t n4, int relu,
                            float4* outF, v4bf* __restrict__ outB) {
  size_t i = (size_t)blockIdx.x * blockDim.x + threadIdx.x;
  if (i >= n4) return;
  int c = (int)(i % M4) * 4;
  float4 h = H[i];
  float4 o;
  o.x = (h.x - mean[c + 0]) * rsig[c + 0] * g[c + 0] + bt[c + 0];
  o.y = (h.y - mean[c + 1]) * rsig[c + 1] * g[c + 1] + bt[c + 1];
  o.z = (h.z - mean[c + 2]) * rsig[c + 2] * g[c + 2] + bt[c + 2];
  o.w = (h.w - mean[c + 3]) * rsig[c + 3] * g[c + 3] + bt[c + 3];
  if (relu) {
    o.x = fmaxf(o.x, 0.f); o.y = fmaxf(o.y, 0.f);
    o.z = fmaxf(o.z, 0.f); o.w = fmaxf(o.w, 0.f);
  }
  outF[i] = o;
  if (outB) {
    v4bf b;
    b[0] = f2b(o.x); b[1] = f2b(o.y); b[2] = f2b(o.z); b[3] = f2b(o.w);
    outB[i] = b;
  }
}

// ==================================================================
extern "C" void kernel_launch(void* const* d_in, const int* in_sizes, int n_in,
                              void* d_out, int out_size, void* d_ws, size_t ws_size,
                              hipStream_t stream) {
  (void)n_in; (void)out_size; (void)ws_size;
  const int N = 40000, F_IN = 128, F_HID = 256, F_OUT = 128;
  const int E = in_sizes[1] / 2;

  const float* x   = (const float*)d_in[0];
  const int*   ei  = (const int*)d_in[1];
  const int*   src = ei;
  const int*   dst = ei + E;
  const float* W1l = (const float*)d_in[2];
  const float* b1  = (const float*)d_in[3];
  const float* W1r = (const float*)d_in[4];
  const float* g1  = (const float*)d_in[5];
  const float* bt1 = (const float*)d_in[6];
  const float* W2l = (const float*)d_in[7];
  const float* b2  = (const float*)d_in[8];
  const float* W2r = (const float*)d_in[9];
  const float* g2  = (const float*)d_in[10];
  const float* bt2 = (const float*)d_in[11];
  float* out = (float*)d_out;

  // ---- workspace carving (256B aligned; deg|agg1|agg2 are exactly contiguous for a one-shot zero)
  char* wsp = (char*)d_ws;
  size_t off = 0;
  auto carve = [&](size_t bytes) -> char* {
    char* p = wsp + off;
    off += (bytes + 255) & ~(size_t)255;
    return p;
  };
  float*  deg   = (float*)carve((size_t)N * 4);
  float*  agg1  = (float*)carve((size_t)N * F_IN * 4);
  float*  agg2  = (float*)carve((size_t)N * F_HID * 4);
  float*  h1    = (float*)carve((size_t)N * F_HID * 4);
  __bf16* xb    = (__bf16*)carve((size_t)N * F_IN * 2);
  __bf16* agg1b = (__bf16*)carve((size_t)N * F_IN * 2);
  __bf16* h1b   = (__bf16*)carve((size_t)N * F_HID * 2);
  __bf16* agg2b = (__bf16*)carve((size_t)N * F_HID * 2);
  __bf16* w1lb  = (__bf16*)carve((size_t)F_HID * F_IN * 2);
  __bf16* w1rb  = (__bf16*)carve((size_t)F_HID * F_IN * 2);
  __bf16* w2lb  = (__bf16*)carve((size_t)F_OUT * F_HID * 2);
  __bf16* w2rb  = (__bf16*)carve((size_t)F_OUT * F_HID * 2);
  float*  mean1 = (float*)carve((size_t)F_HID * 4);
  float*  rs1   = (float*)carve((size_t)F_HID * 4);
  float*  mean2 = (float*)carve((size_t)F_OUT * 4);
  float*  rs2   = (float*)carve((size_t)F_OUT * 4);

  const int T = 256;
  auto nb = [](size_t n, int t) { return (unsigned)((n + (size_t)t - 1) / t); };

  // 1) zero accumulators (deg + agg1 + agg2, contiguous)
  {
    size_t z4 = (size_t)N * (1 + F_IN + F_HID) / 4;
    k_zero4<<<nb(z4, T), T, 0, stream>>>((float4*)deg, z4);
  }
  // 2) bf16 conversions: x and all 4 weight matrices
  {
    size_t n4 = (size_t)N * F_IN / 4;
    k_cvt_bf16_4<<<nb(n4, T), T, 0, stream>>>((const float4*)x, (v4bf*)xb, n4);
    size_t w14 = (size_t)F_HID * F_IN / 4;
    k_cvt_bf16_4<<<nb(w14, T), T, 0, stream>>>((const float4*)W1l, (v4bf*)w1lb, w14);
    k_cvt_bf16_4<<<nb(w14, T), T, 0, stream>>>((const float4*)W1r, (v4bf*)w1rb, w14);
    size_t w24 = (size_t)F_OUT * F_HID / 4;
    k_cvt_bf16_4<<<nb(w24, T), T, 0, stream>>>((const float4*)W2l, (v4bf*)w2lb, w24);
    k_cvt_bf16_4<<<nb(w24, T), T, 0, stream>>>((const float4*)W2r, (v4bf*)w2rb, w24);
  }
  // 3) degrees
  k_degree<<<nb((size_t)E, T), T, 0, stream>>>(dst, deg, E);

  // ================= layer 1 =================
  {
    long long tot = (long long)E * (F_IN / 4);
    k_scatter4<<<nb((size_t)tot, T), T, 0, stream>>>(src, dst, x, agg1, F_IN / 4, F_IN, tot);
    size_t n4 = (size_t)N * F_IN / 4;
    k_mean_bf16<<<nb(n4, T), T, 0, stream>>>((const float4*)agg1, deg, (v4bf*)agg1b, F_IN / 4, n4);
    int tiles = (N / 16) * (F_HID / 16);            // 40000 tiles
    k_sage_gemm<<<(unsigned)(tiles / 8), 256, 0, stream>>>(
        agg1b, xb, w1lb, w1rb, b1, h1, F_IN, F_HID, F_HID / 16, tiles);
    k_bn_stats<<<F_HID, 256, 0, stream>>>(h1, N, F_HID, mean1, rs1);
    size_t h4 = (size_t)N * F_HID / 4;
    k_bn_apply4<<<nb(h4, T), T, 0, stream>>>((const float4*)h1, mean1, rs1, g1, bt1,
                                             F_HID / 4, h4, 1, (float4*)h1, (v4bf*)h1b);
  }
  // ================= layer 2 =================
  {
    long long tot = (long long)E * (F_HID / 4);
    k_scatter4<<<nb((size_t)tot, T), T, 0, stream>>>(src, dst, h1, agg2, F_HID / 4, F_HID, tot);
    size_t n4 = (size_t)N * F_HID / 4;
    k_mean_bf16<<<nb(n4, T), T, 0, stream>>>((const float4*)agg2, deg, (v4bf*)agg2b, F_HID / 4, n4);
    int tiles = (N / 16) * (F_OUT / 16);            // 20000 tiles
    k_sage_gemm<<<(unsigned)(tiles / 8), 256, 0, stream>>>(
        agg2b, h1b, w2lb, w2rb, b2, out, F_HID, F_OUT, F_OUT / 16, tiles);
    k_bn_stats<<<F_OUT, 256, 0, stream>>>(out, N, F_OUT, mean2, rs2);
    size_t o4 = (size_t)N * F_OUT / 4;
    k_bn_apply4<<<nb(o4, T), T, 0, stream>>>((const float4*)out, mean2, rs2, g2, bt2,
                                             F_OUT / 4, o4, 0, (float4*)out, (v4bf*)nullptr);
  }
}
